// DoubleAttentionLayer_50190987821494
// MI455X (gfx1250) — matrix operand order repaired and checked by reading, not compiled, not run on hardware
//
#include <hip/hip_runtime.h>
#include <hip/hip_bf16.h>

// Problem constants (from reference): B=2, C=128, H=W=D=48, c_m=128, c_n=64
#define B_   2
#define C_   128
#define CM_  128
#define CN_  64
#define L_   110592          // 48*48*48
#define ROWS_ 256            // CM_ + CN_ + CN_
#define LT_  128             // L-columns per workgroup in k1/k3/k4
#define EPS_ 1e-5f

typedef __bf16 bf16;
typedef __attribute__((ext_vector_type(16))) __bf16 v16bf;
typedef __attribute__((ext_vector_type(8)))  float  v8f;

__device__ __forceinline__ bf16 f2bf(float f) {
  unsigned u = __float_as_uint(f);
  unsigned short h = (unsigned short)((u + 0x7FFFu + ((u >> 16) & 1u)) >> 16);
  bf16 r; __builtin_memcpy(&r, &h, 2); return r;
}

// ---- WMMA fragment builders (contiguous k order, same for A and B, wave32) ----
// A-fragment 16x32 bf16: lane holds row m = lane&15; half i of v16bf:
//   K = 16*(lane>>4) + i
// (this gather is one contiguous 32B chunk per lane -> compiler emits ds_load_b128 x2)
__device__ __forceinline__ v16bf load_a_frag(const bf16* base, int stride, int lane) {
  v16bf a; const int m = lane & 15; const int khi = (lane >> 4) << 4;
  #pragma unroll
  for (int i = 0; i < 16; ++i) {
    int k = khi + i;
    a[i] = base[m * stride + k];
  }
  return a;
}
// B-fragment 32x16 bf16 from a [k][n] (row-major in k) source:
//   lane holds col n = lane&15; half i: K = 16*(lane>>4) + i
__device__ __forceinline__ v16bf load_b_frag(const bf16* base, int kstride, int lane) {
  v16bf b; const int n = lane & 15; const int k0 = (lane >> 4) << 4;
  #pragma unroll
  for (int i = 0; i < 16; ++i) b[i] = base[(k0 + i) * kstride + n];
  return b;
}
// B-fragment 32x16 bf16 from an [n][k] (transposed) source
__device__ __forceinline__ v16bf load_b_frag_T(const bf16* base, int nstride, int lane) {
  v16bf b; const int n = lane & 15; const int k0 = (lane >> 4) << 4;
  #pragma unroll
  for (int i = 0; i < 16; ++i) b[i] = base[n * nstride + k0 + i];
  return b;
}

#define WMMA_BF16(a, b, c) \
  __builtin_amdgcn_wmma_f32_16x16x32_bf16(false, (a), false, (b), (short)0, (c), false, false)

// ---------------------------------------------------------------------------
// k0: zero atomic accumulators (workspace is poisoned 0xAA by harness)
// ---------------------------------------------------------------------------
__global__ void k0_zero(float* __restrict__ G, float* __restrict__ macc,
                        float* __restrict__ qacc) {
  int t = threadIdx.x + blockIdx.x * blockDim.x;
  for (int i = t; i < B_ * CM_ * CN_; i += blockDim.x * gridDim.x) G[i] = 0.f;
  if (t < B_ * CM_) { macc[t] = 0.f; qacc[t] = 0.f; }
}

// ---------------------------------------------------------------------------
// k1: fused projection  Y(256,L) = W(256,128) @ X(128,L) + bias
//     rows 0..127 -> A (stored bf16), 128..191 -> Bm (f32), 192..255 -> V (f32)
//     Row-tile loop split into three fixed-destination loops: no runtime
//     branches or global bias loads in the hot path.
// ---------------------------------------------------------------------------
__global__ void k1_proj(const float* __restrict__ x,
                        const float* __restrict__ WA, const float* __restrict__ bA,
                        const float* __restrict__ WB, const float* __restrict__ bB,
                        const float* __restrict__ WV, const float* __restrict__ bV,
                        bf16* __restrict__ Abuf, float* __restrict__ Bmbuf,
                        float* __restrict__ Vbuf) {
  __shared__ bf16  Wl[ROWS_ * C_];         // 64 KB: combined weights, bf16
  __shared__ float biasl[ROWS_];           // 1 KB: combined biases
  const int tid = threadIdx.x;
  const int b   = blockIdx.y;
  const int l0blk = blockIdx.x * LT_;

  for (int j = tid; j < ROWS_ * C_; j += 256) {
    int row = j >> 7, c = j & 127;
    float w = (row < CM_)        ? WA[j]
            : (row < CM_ + CN_)  ? WB[(row - CM_) * C_ + c]
                                 : WV[(row - CM_ - CN_) * C_ + c];
    Wl[j] = f2bf(w);
  }
  if (tid < ROWS_) {
    biasl[tid] = (tid < CM_)       ? bA[tid]
               : (tid < CM_ + CN_) ? bB[tid - CM_]
                                   : bV[tid - CM_ - CN_];
  }
  __syncthreads();

  const int wave = tid >> 5, lane = tid & 31;
  const int l0  = l0blk + wave * 16;       // this wave's 16-column tile
  const int col = l0 + (lane & 15);
  const int rowoff = (lane >> 4) << 3;     // C/D row offset for this half-wave

  // B-fragments for K = 0..127 over this column tile, held in registers
  v16bf bfr[4];
  #pragma unroll
  for (int kk = 0; kk < 4; ++kk) {
    const int kbase = kk * 32 + ((lane >> 4) << 4);
    #pragma unroll
    for (int i = 0; i < 16; ++i)
      bfr[kk][i] = f2bf(x[(size_t)(b * C_ + kbase + i) * L_ + col]);
  }

  bf16*  aout = Abuf  + (size_t)b * CM_ * L_ + col;
  float* bout = Bmbuf + (size_t)b * CN_ * L_ + col;
  float* vout = Vbuf  + (size_t)b * CN_ * L_ + col;

  // rows 0..127 -> A (bf16)
  for (int rt = 0; rt < 8; ++rt) {
    v8f acc = {0.f, 0.f, 0.f, 0.f, 0.f, 0.f, 0.f, 0.f};
    #pragma unroll
    for (int kk = 0; kk < 4; ++kk) {
      v16bf a = load_a_frag(&Wl[(rt * 16) * C_ + kk * 32], C_, lane);
      acc = WMMA_BF16(a, bfr[kk], acc);
    }
    #pragma unroll
    for (int r = 0; r < 8; ++r) {
      int row = rt * 16 + r + rowoff;
      aout[(size_t)row * L_] = f2bf(acc[r] + biasl[row]);
    }
  }
  // rows 128..191 -> Bm (f32)
  for (int rt = 8; rt < 12; ++rt) {
    v8f acc = {0.f, 0.f, 0.f, 0.f, 0.f, 0.f, 0.f, 0.f};
    #pragma unroll
    for (int kk = 0; kk < 4; ++kk) {
      v16bf a = load_a_frag(&Wl[(rt * 16) * C_ + kk * 32], C_, lane);
      acc = WMMA_BF16(a, bfr[kk], acc);
    }
    #pragma unroll
    for (int r = 0; r < 8; ++r) {
      int row = rt * 16 + r + rowoff;
      bout[(size_t)(row - CM_) * L_] = acc[r] + biasl[row];
    }
  }
  // rows 192..255 -> V (f32)
  for (int rt = 12; rt < 16; ++rt) {
    v8f acc = {0.f, 0.f, 0.f, 0.f, 0.f, 0.f, 0.f, 0.f};
    #pragma unroll
    for (int kk = 0; kk < 4; ++kk) {
      v16bf a = load_a_frag(&Wl[(rt * 16) * C_ + kk * 32], C_, lane);
      acc = WMMA_BF16(a, bfr[kk], acc);
    }
    #pragma unroll
    for (int r = 0; r < 8; ++r) {
      int row = rt * 16 + r + rowoff;
      vout[(size_t)(row - CM_ - CN_) * L_] = acc[r] + biasl[row];
    }
  }
}

// ---------------------------------------------------------------------------
// k2: per-(b,n) softmax statistics over L: rowmax and 1/sum(exp(x-max))
//     float4 scans (L_ % (256*4) == 0)
// ---------------------------------------------------------------------------
__global__ void k2_rowstats(const float* __restrict__ Bm, float* __restrict__ rmax,
                            float* __restrict__ rsuminv) {
  __shared__ float red[256];
  const int row = blockIdx.x;              // 0 .. B_*CN_-1
  const float* p = Bm + (size_t)row * L_;
  const int t = threadIdx.x;

  float mx = -3.4e38f;
  for (int i = t * 4; i < L_; i += 1024) {
    float4 v = *reinterpret_cast<const float4*>(p + i);
    mx = fmaxf(mx, fmaxf(fmaxf(v.x, v.y), fmaxf(v.z, v.w)));
  }
  red[t] = mx; __syncthreads();
  for (int s = 128; s > 0; s >>= 1) {
    if (t < s) red[t] = fmaxf(red[t], red[t + s]);
    __syncthreads();
  }
  mx = red[0]; __syncthreads();

  float sum = 0.f;
  for (int i = t * 4; i < L_; i += 1024) {
    float4 v = *reinterpret_cast<const float4*>(p + i);
    sum += __expf(v.x - mx) + __expf(v.y - mx) + __expf(v.z - mx) + __expf(v.w - mx);
  }
  red[t] = sum; __syncthreads();
  for (int s = 128; s > 0; s >>= 1) {
    if (t < s) red[t] += red[t + s];
    __syncthreads();
  }
  if (t == 0) { rmax[row] = mx; rsuminv[row] = 1.f / red[0]; }
}

// ---------------------------------------------------------------------------
// k3: G[b,m,n] = sum_l A[b,m,l] * softmax_l(Bm)[b,n,l]   (K-split + atomics)
// ---------------------------------------------------------------------------
__global__ void k3_gmat(const bf16* __restrict__ Abuf, const float* __restrict__ Bmbuf,
                        const float* __restrict__ rmax, const float* __restrict__ rsuminv,
                        float* __restrict__ G) {
  __shared__ bf16 Al[CM_ * LT_];           // 32 KB
  __shared__ bf16 Pl[CN_ * LT_];           // 16 KB
  const int tid = threadIdx.x;
  const int b   = blockIdx.y;
  const int l0  = blockIdx.x * LT_;

  // A chunk: uint4 = 8 bf16 per load/store (row = 128 bf16 = 16 uint4)
  {
    uint4* dst = reinterpret_cast<uint4*>(Al);
    for (int j = tid; j < (CM_ * LT_) / 8; j += 256) {
      int m = j >> 4, c8 = j & 15;
      const uint4* src = reinterpret_cast<const uint4*>(
          Abuf + (size_t)(b * CM_ + m) * L_ + l0);
      dst[j] = src[c8];
    }
  }
  // P chunk: float4 reads of Bm, exponentiate, scale, store bf16
  for (int j = tid; j < (CN_ * LT_) / 4; j += 256) {
    int n = j >> 5, c4 = j & 31;
    float4 v = *reinterpret_cast<const float4*>(
        Bmbuf + (size_t)(b * CN_ + n) * L_ + l0 + c4 * 4);
    float mx  = rmax[b * CN_ + n];
    float inv = rsuminv[b * CN_ + n];
    bf16* q = &Pl[n * LT_ + c4 * 4];
    q[0] = f2bf(__expf(v.x - mx) * inv);
    q[1] = f2bf(__expf(v.y - mx) * inv);
    q[2] = f2bf(__expf(v.z - mx) * inv);
    q[3] = f2bf(__expf(v.w - mx) * inv);
  }
  __syncthreads();

  const int wave = tid >> 5, lane = tid & 31;
  const int m0 = wave * 16;
  #pragma unroll
  for (int nt = 0; nt < 4; ++nt) {
    v8f acc = {0.f, 0.f, 0.f, 0.f, 0.f, 0.f, 0.f, 0.f};
    #pragma unroll
    for (int ks = 0; ks < 4; ++ks) {
      v16bf a  = load_a_frag(&Al[m0 * LT_ + ks * 32], LT_, lane);
      v16bf bb = load_b_frag_T(&Pl[(nt * 16) * LT_ + ks * 32], LT_, lane);
      acc = WMMA_BF16(a, bb, acc);
    }
    #pragma unroll
    for (int r = 0; r < 8; ++r) {
      int m = m0 + r + ((lane >> 4) << 3);
      int n = nt * 16 + (lane & 15);
      atomicAdd(&G[(b * CM_ + m) * CN_ + n], acc[r]);
    }
  }
}

// ---------------------------------------------------------------------------
// k4: Z[b,m,l] = sum_n G[b,m,n] * softmax_n(V)[b,n,l]; also Σ and Σ² per (b,m)
// ---------------------------------------------------------------------------
__global__ void k4_zmat(const float* __restrict__ G, const float* __restrict__ Vbuf,
                        float* __restrict__ Z, float* __restrict__ macc,
                        float* __restrict__ qacc) {
  __shared__ bf16 Gl[CM_ * CN_];           // 16 KB
  __shared__ bf16 Pl[CN_ * LT_];           // 16 KB
  const int tid = threadIdx.x;
  const int b   = blockIdx.y;
  const int l0  = blockIdx.x * LT_;

  for (int j = tid; j < (CM_ * CN_) / 4; j += 256) {
    float4 g = reinterpret_cast<const float4*>(G + (size_t)b * CM_ * CN_)[j];
    bf16* q = &Gl[j * 4];
    q[0] = f2bf(g.x); q[1] = f2bf(g.y); q[2] = f2bf(g.z); q[3] = f2bf(g.w);
  }

  if (tid < LT_) {                          // per-voxel channel softmax over n=64
    const int lc = tid;
    float mx = -3.4e38f;
    for (int n = 0; n < CN_; ++n)
      mx = fmaxf(mx, Vbuf[(size_t)(b * CN_ + n) * L_ + l0 + lc]);
    float s = 0.f;
    for (int n = 0; n < CN_; ++n) {
      float e = __expf(Vbuf[(size_t)(b * CN_ + n) * L_ + l0 + lc] - mx);
      s += e;
      Pl[n * LT_ + lc] = f2bf(e);
    }
    float inv = 1.f / s;
    for (int n = 0; n < CN_; ++n)
      Pl[n * LT_ + lc] = f2bf((float)Pl[n * LT_ + lc] * inv);
  }
  __syncthreads();

  const int wave = tid >> 5, lane = tid & 31;
  const int m0 = wave * 16;
  v16bf a0 = load_a_frag(&Gl[m0 * CN_ + 0],  CN_, lane);
  v16bf a1 = load_a_frag(&Gl[m0 * CN_ + 32], CN_, lane);

  float zs[8] = {0, 0, 0, 0, 0, 0, 0, 0};
  float zq[8] = {0, 0, 0, 0, 0, 0, 0, 0};
  #pragma unroll 2
  for (int ct = 0; ct < 8; ++ct) {
    v8f acc = {0.f, 0.f, 0.f, 0.f, 0.f, 0.f, 0.f, 0.f};
    v16bf b0 = load_b_frag(&Pl[0 * LT_  + ct * 16], LT_, lane);
    v16bf b1 = load_b_frag(&Pl[32 * LT_ + ct * 16], LT_, lane);
    acc = WMMA_BF16(a0, b0, acc);
    acc = WMMA_BF16(a1, b1, acc);
    const int col = l0 + ct * 16 + (lane & 15);
    #pragma unroll
    for (int r = 0; r < 8; ++r) {
      int m = m0 + r + ((lane >> 4) << 3);
      float v = acc[r];
      Z[(size_t)(b * CM_ + m) * L_ + col] = v;
      zs[r] += v; zq[r] += v * v;
    }
  }
  // reduce across the 16 lanes that share a row, then one atomic per row
  #pragma unroll
  for (int r = 0; r < 8; ++r) {
    float s = zs[r], q = zq[r];
    #pragma unroll
    for (int msk = 1; msk <= 8; msk <<= 1) {
      s += __shfl_xor(s, msk, 32);
      q += __shfl_xor(q, msk, 32);
    }
    if ((lane & 15) == 0) {
      int m = m0 + r + ((lane >> 4) << 3);
      atomicAdd(&macc[b * CM_ + m], s);
      atomicAdd(&qacc[b * CM_ + m], q);
    }
  }
}

// ---------------------------------------------------------------------------
// k5: in-place InstanceNorm finalize (biased variance), float4 vectorized
// ---------------------------------------------------------------------------
__global__ void k5_norm(float* __restrict__ Z, const float* __restrict__ macc,
                        const float* __restrict__ qacc) {
  const size_t idx = (size_t)blockIdx.x * blockDim.x + threadIdx.x;
  const size_t e = idx * 4;
  if (e >= (size_t)B_ * CM_ * L_) return;
  const int bm = (int)(e / L_);             // L_ % 4 == 0 -> all 4 share (b,m)
  const float invL = 1.f / (float)L_;
  float mean  = macc[bm] * invL;
  float var   = qacc[bm] * invL - mean * mean;
  float scale = rsqrtf(var + EPS_);
  float4 z = *reinterpret_cast<float4*>(Z + e);
  z.x = (z.x - mean) * scale; z.y = (z.y - mean) * scale;
  z.z = (z.z - mean) * scale; z.w = (z.w - mean) * scale;
  *reinterpret_cast<float4*>(Z + e) = z;
}

// ---------------------------------------------------------------------------
extern "C" void kernel_launch(void* const* d_in, const int* in_sizes, int n_in,
                              void* d_out, int out_size, void* d_ws, size_t ws_size,
                              hipStream_t stream) {
  const float* x  = (const float*)d_in[0];
  const float* WA = (const float*)d_in[1];
  const float* bA = (const float*)d_in[2];
  const float* WB = (const float*)d_in[3];
  const float* bB = (const float*)d_in[4];
  const float* WV = (const float*)d_in[5];
  const float* bV = (const float*)d_in[6];
  float* out = (float*)d_out;

  char* ws = (char*)d_ws;
  size_t off = 0;
  auto wsalloc = [&](size_t bytes) -> char* {
    char* p = ws + off;
    off += (bytes + 255) & ~(size_t)255;
    return p;
  };
  bf16*  Abuf    = (bf16*) wsalloc((size_t)B_ * CM_ * L_ * sizeof(bf16));   // ~56.6 MB
  float* Bmbuf   = (float*)wsalloc((size_t)B_ * CN_ * L_ * sizeof(float));  // ~56.6 MB
  float* Vbuf    = (float*)wsalloc((size_t)B_ * CN_ * L_ * sizeof(float));  // ~56.6 MB
  float* rmax    = (float*)wsalloc((size_t)B_ * CN_ * sizeof(float));
  float* rsuminv = (float*)wsalloc((size_t)B_ * CN_ * sizeof(float));
  float* G       = (float*)wsalloc((size_t)B_ * CM_ * CN_ * sizeof(float));
  float* macc    = (float*)wsalloc((size_t)B_ * CM_ * sizeof(float));
  float* qacc    = (float*)wsalloc((size_t)B_ * CM_ * sizeof(float));

  const dim3 blk(256, 1, 1);
  k0_zero    <<<dim3(64),            blk, 0, stream>>>(G, macc, qacc);
  k1_proj    <<<dim3(L_ / LT_, B_),  blk, 0, stream>>>(x, WA, bA, WB, bB, WV, bV,
                                                       Abuf, Bmbuf, Vbuf);
  k2_rowstats<<<dim3(B_ * CN_),      blk, 0, stream>>>(Bmbuf, rmax, rsuminv);
  k3_gmat    <<<dim3(L_ / LT_, B_),  blk, 0, stream>>>(Abuf, Bmbuf, rmax, rsuminv, G);
  k4_zmat    <<<dim3(L_ / LT_, B_),  blk, 0, stream>>>(G, Vbuf, out, macc, qacc);
  k5_norm    <<<dim3((B_ * CM_ * L_ / 4 + 255) / 256), blk, 0, stream>>>(out, macc, qacc);
}
